// SGModifiedSimpleViT_12584254177690
// MI455X (gfx1250) — matrix operand order repaired and checked by reading, not compiled, not run on hardware
//
#include <hip/hip_runtime.h>
#include <hip/hip_bf16.h>

typedef __attribute__((ext_vector_type(16))) _Float16 v16h;
typedef __attribute__((ext_vector_type(8)))  _Float16 v8h;
typedef __attribute__((ext_vector_type(8)))  float    v8f;

#define BATCH 8
#define CHN 6
#define IMGSZ 384
#define PSZ 16
#define DIM 768
#define HEADS 12
#define DEPTH 4
#define MLPD 3072
#define NCLS 1000
#define NTOK 576            // (384/16)^2
#define DHD 64
#define PATCH_DIM 1536
#define TOKS (BATCH * NTOK) // 4608
#define GM 2                // M-subtiles (16 rows each) per GEMM wave
#define GN 4                // N-subtiles (16 cols each) per GEMM wave

// ---- fragment load helpers (two 16B loads each; all addresses 16B aligned) ----
static __device__ inline v16h cat8(v8h lo, v8h hi) {
  return __builtin_shufflevector(lo, hi, 0, 1, 2, 3, 4, 5, 6, 7,
                                 8, 9, 10, 11, 12, 13, 14, 15);
}
// A fragment: 16 contiguous halves at p
static __device__ inline v16h load_frag_a(const _Float16* p) {
  return cat8(*(const v8h*)p, *(const v8h*)(p + 8));
}
// B fragment: 16 contiguous halves at p
static __device__ inline v16h load_frag_b(const _Float16* p) {
  return cat8(*(const v8h*)p, *(const v8h*)(p + 8));
}

// ---------------------------------------------------------------------------
// im2col: x[b][c][h][w] -> Apatch f16 [TOKS][PATCH_DIM], col = (p1*16+p2)*6 + c
// ---------------------------------------------------------------------------
__global__ void im2col_kernel(const float* __restrict__ x, _Float16* __restrict__ Ap) {
  size_t idx = (size_t)blockIdx.x * blockDim.x + threadIdx.x;
  const size_t total = (size_t)TOKS * PATCH_DIM;
  if (idx >= total) return;
  int col = (int)(idx % PATCH_DIM);
  int row = (int)(idx / PATCH_DIM);
  int c  = col % CHN;
  int pp = col / CHN;
  int p2 = pp % PSZ, p1 = pp / PSZ;
  int gw = row % 24; int t = row / 24;
  int gh = t % 24;   int b = t / 24;
  int hh = gh * PSZ + p1, ww = gw * PSZ + p2;
  Ap[idx] = (_Float16)x[(((size_t)b * CHN + c) * IMGSZ + hh) * IMGSZ + ww];
}

// ---------------------------------------------------------------------------
// convert + transpose: W f32 [K][N] -> Wt f16 [N][K]
// ---------------------------------------------------------------------------
__global__ void convt_kernel(const float* __restrict__ W, _Float16* __restrict__ Wt,
                             int K, int N) {
  size_t idx = (size_t)blockIdx.x * blockDim.x + threadIdx.x;
  size_t total = (size_t)K * N;
  if (idx >= total) return;
  int k = (int)(idx / N), n = (int)(idx % N);
  Wt[(size_t)n * K + k] = (_Float16)W[idx];
}

// ---------------------------------------------------------------------------
// V transpose per layer: vt[b][h][d][n] = qkv[(b*576+n)*2304 + 1536 + h*64 + d]
// ---------------------------------------------------------------------------
__global__ void vtrans_kernel(const _Float16* __restrict__ qkv, _Float16* __restrict__ vt) {
  size_t idx = (size_t)blockIdx.x * blockDim.x + threadIdx.x;
  const size_t total = (size_t)BATCH * HEADS * DHD * NTOK;
  if (idx >= total) return;
  int n = (int)(idx % NTOK); size_t t = idx / NTOK;
  int d = (int)(t % DHD);    t /= DHD;
  int h = (int)(t % HEADS);
  int b = (int)(t / HEADS);
  vt[idx] = qkv[(size_t)(b * NTOK + n) * (3 * DIM) + 2 * DIM + h * DHD + d];
}

// ---------------------------------------------------------------------------
// per-layer log-softmax of attention bias rows (length 576)
// ---------------------------------------------------------------------------
__global__ void bias_logsoftmax_kernel(const float* __restrict__ src_all,
                                       float* __restrict__ dst) {
  int row = blockIdx.x;            // [0, 12*576)
  const float* src = src_all + (size_t)row * NTOK;
  float* out = dst + (size_t)row * NTOK;
  __shared__ float red[256];
  int tid = threadIdx.x;
  float mx = -1e30f;
  for (int j = tid; j < NTOK; j += 256) mx = fmaxf(mx, src[j]);
  red[tid] = mx; __syncthreads();
  for (int s = 128; s > 0; s >>= 1) { if (tid < s) red[tid] = fmaxf(red[tid], red[tid + s]); __syncthreads(); }
  mx = red[0]; __syncthreads();
  float sum = 0.f;
  for (int j = tid; j < NTOK; j += 256) sum += expf(src[j] - mx);
  red[tid] = sum; __syncthreads();
  for (int s = 128; s > 0; s >>= 1) { if (tid < s) red[tid] += red[tid + s]; __syncthreads(); }
  float lse = mx + logf(red[0]);
  for (int j = tid; j < NTOK; j += 256) out[j] = src[j] - lse;
}

// ---------------------------------------------------------------------------
// LayerNorm rows of X f32 [T][D] -> Y f16
// ---------------------------------------------------------------------------
__global__ void layernorm_kernel(const float* __restrict__ X, const float* __restrict__ g,
                                 const float* __restrict__ bt, _Float16* __restrict__ Y, int D) {
  int row = blockIdx.x;
  const float* xr = X + (size_t)row * D;
  __shared__ float red[256];
  int tid = threadIdx.x;
  float s = 0.f;
  for (int j = tid; j < D; j += 256) s += xr[j];
  red[tid] = s; __syncthreads();
  for (int st = 128; st > 0; st >>= 1) { if (tid < st) red[tid] += red[tid + st]; __syncthreads(); }
  float mean = red[0] / D; __syncthreads();
  float v = 0.f;
  for (int j = tid; j < D; j += 256) { float d = xr[j] - mean; v += d * d; }
  red[tid] = v; __syncthreads();
  for (int st = 128; st > 0; st >>= 1) { if (tid < st) red[tid] += red[tid + st]; __syncthreads(); }
  float rstd = rsqrtf(red[0] / D + 1e-5f);
  for (int j = tid; j < D; j += 256)
    Y[(size_t)row * D + j] = (_Float16)((xr[j] - mean) * rstd * g[j] + bt[j]);
}

// ---------------------------------------------------------------------------
// mean pool over tokens: tok f32 [B][576][768] -> pooled f32 [B][768]
// ---------------------------------------------------------------------------
__global__ void pool_kernel(const float* __restrict__ tok, float* __restrict__ pooled) {
  int b = blockIdx.x;
  for (int d = threadIdx.x; d < DIM; d += 256) {
    float s = 0.f;
    for (int n = 0; n < NTOK; ++n) s += tok[(size_t)(b * NTOK + n) * DIM + d];
    pooled[(size_t)b * DIM + d] = s * (1.0f / NTOK);
  }
}

// ---------------------------------------------------------------------------
// WMMA GEMM: C[M][N] = A[M][K](f16) * Bt[N][K](f16)^T  (+bias, gelu, residual)
// One wave computes a (16*GM)x(16*GN) tile: per K-step, 6 fragment loads
// (12x b128) feed 8 WMMAs. K-loop kept at unroll(1) so the live set
// (64 acc + 48 frag VGPRs) stays spill-free; latency is hidden by occupancy.
// Out-of-range rows/cols are CLAMPED on load; their products land in
// never-stored C elements, keeping the WMMA loop branch-free with full EXEC.
// ---------------------------------------------------------------------------
__global__ void gemm_wmma_kernel(const _Float16* __restrict__ A,
                                 const _Float16* __restrict__ Bt,
                                 const float* __restrict__ bias,
                                 const float* __restrict__ residual,
                                 float* __restrict__ Cf,
                                 _Float16* __restrict__ Ch,
                                 int M, int N, int K, int act_gelu) {
  int lane = threadIdx.x;
  int n0 = blockIdx.x * (16 * GN);
  int m0 = blockIdx.y * (16 * GM);
  int lrow = lane & 15;
  int hi = (lane >> 4) & 1;

  const _Float16* Ar[GM];
#pragma unroll
  for (int i = 0; i < GM; ++i) {
    int arow = m0 + i * 16 + lrow;
    int arc = arow < M ? arow : M - 1;
    Ar[i] = A + (size_t)arc * K + (hi ? 16 : 0);
  }

  int bcol[GN];
  const _Float16* Br[GN];
#pragma unroll
  for (int t = 0; t < GN; ++t) {
    bcol[t] = n0 + t * 16 + lrow;
    int bc = bcol[t] < N ? bcol[t] : N - 1;
    Br[t] = Bt + (size_t)bc * K + (hi ? 16 : 0);
  }

  v8f acc[GM][GN];
#pragma unroll
  for (int i = 0; i < GM; ++i)
#pragma unroll
    for (int t = 0; t < GN; ++t)
#pragma unroll
      for (int r = 0; r < 8; ++r) acc[i][t][r] = 0.f;

#pragma unroll 1
  for (int k0 = 0; k0 < K; k0 += 32) {
    v16h a[GM], bf[GN];
#pragma unroll
    for (int i = 0; i < GM; ++i) a[i] = load_frag_a(Ar[i] + k0);
#pragma unroll
    for (int t = 0; t < GN; ++t) bf[t] = load_frag_b(Br[t] + k0);
#pragma unroll
    for (int i = 0; i < GM; ++i)
#pragma unroll
      for (int t = 0; t < GN; ++t)
        acc[i][t] = __builtin_amdgcn_wmma_f32_16x16x32_f16(false, a[i], false, bf[t],
                                                           (short)0, acc[i][t], false, false);
  }

  // --- epilogue ---
#pragma unroll
  for (int t = 0; t < GN; ++t) {
    if (bcol[t] >= N) continue;
    float bv = bias ? bias[bcol[t]] : 0.f;
#pragma unroll
    for (int i = 0; i < GM; ++i) {
#pragma unroll
      for (int r = 0; r < 8; ++r) {
        int m = m0 + i * 16 + r + hi * 8;
        if (m >= M) continue;
        float v = acc[i][t][r] + bv;
        if (act_gelu) v = 0.5f * v * (1.f + erff(v * 0.70710678118654752f));
        size_t off = (size_t)m * N + bcol[t];
        if (residual) v += residual[off];
        if (Cf) Cf[off] = v;
        if (Ch) Ch[off] = (_Float16)v;
      }
    }
  }
}

// ---------------------------------------------------------------------------
// Flash attention with additive log-softmax(bias).
// attn = softmax(Q*K^T*scale + logsb); O = attn * V.  One wave per
// (16-query tile, head, batch). qkv f16 [TOKS][2304], vt f16 [B][H][64][576].
// ---------------------------------------------------------------------------
__global__ void attn_kernel(const _Float16* __restrict__ qkv,
                            const _Float16* __restrict__ vt,
                            const float* __restrict__ sblog,
                            _Float16* __restrict__ obuf) {
  int lane = threadIdx.x;
  int i0 = blockIdx.x * 16;
  int h  = blockIdx.y;
  int b  = blockIdx.z;
  int lrow = lane & 15;
  int hi = (lane >> 4) & 1;

  const _Float16* Qp = qkv + (size_t)(b * NTOK) * (3 * DIM) + h * DHD;
  const _Float16* Kp = Qp + DIM;
  const _Float16* Vt = vt + ((size_t)(b * HEADS + h) * DHD) * NTOK; // [64][576]
  const float* SB = sblog + ((size_t)h * NTOK + i0) * NTOK;

  // Q fragments for the two 32-wide K-dim steps (DH = 64)
  const _Float16* qrow = Qp + (size_t)(i0 + lrow) * (3 * DIM) + (hi ? 16 : 0);
  v16h qa0 = load_frag_a(qrow);
  v16h qa1 = load_frag_a(qrow + 32);

  float mrow[8], lsum[8];
  v8f accO[4];
#pragma unroll
  for (int c = 0; c < 4; ++c)
#pragma unroll
    for (int r = 0; r < 8; ++r) accO[c][r] = 0.f;
#pragma unroll
  for (int r = 0; r < 8; ++r) { mrow[r] = -1e30f; lsum[r] = 0.f; }

  __shared__ __align__(16) _Float16 Pt[16 * 32];
  const float scale = 0.125f; // 1/sqrt(64)

#pragma unroll 1
  for (int kt = 0; kt < NTOK; kt += 32) {
    // ---- S = Q*K^T for two 16-key tiles ----
    v8f st[2];
#pragma unroll
    for (int t = 0; t < 2; ++t) {
#pragma unroll
      for (int r = 0; r < 8; ++r) st[t][r] = 0.f;
      const _Float16* kr = Kp + (size_t)(kt + t * 16 + lrow) * (3 * DIM) + (hi ? 16 : 0);
      st[t] = __builtin_amdgcn_wmma_f32_16x16x32_f16(false, qa0, false, load_frag_b(kr),
                                                     (short)0, st[t], false, false);
      st[t] = __builtin_amdgcn_wmma_f32_16x16x32_f16(false, qa1, false, load_frag_b(kr + 32),
                                                     (short)0, st[t], false, false);
    }
    // ---- logits + online softmax update ----
    float t0[8], t1[8], f[8];
#pragma unroll
    for (int r = 0; r < 8; ++r) {
      int m = r + hi * 8;
      t0[r] = st[0][r] * scale + SB[(size_t)m * NTOK + kt + lrow];
      t1[r] = st[1][r] * scale + SB[(size_t)m * NTOK + kt + 16 + lrow];
    }
#pragma unroll
    for (int r = 0; r < 8; ++r) {
      float mx = fmaxf(t0[r], t1[r]);
      for (int msk = 1; msk < 16; msk <<= 1) mx = fmaxf(mx, __shfl_xor(mx, msk, 32));
      float mnew = fmaxf(mrow[r], mx);
      f[r] = expf(mrow[r] - mnew);
      t0[r] = expf(t0[r] - mnew);
      t1[r] = expf(t1[r] - mnew);
      float rs = t0[r] + t1[r];
      for (int msk = 1; msk < 16; msk <<= 1) rs += __shfl_xor(rs, msk, 32);
      lsum[r] = lsum[r] * f[r] + rs;
      mrow[r] = mnew;
    }
#pragma unroll
    for (int c = 0; c < 4; ++c)
#pragma unroll
      for (int r = 0; r < 8; ++r) accO[c][r] *= f[r];

    // ---- P through LDS: C-layout -> A-layout (f16) ----
    __syncthreads();
#pragma unroll
    for (int r = 0; r < 8; ++r) {
      int m = r + hi * 8;
      Pt[m * 32 + lrow]      = (_Float16)t0[r];
      Pt[m * 32 + 16 + lrow] = (_Float16)t1[r];
    }
    __syncthreads();
    v16h pa = load_frag_a(&Pt[lrow * 32 + (hi ? 16 : 0)]);

    // ---- O += P * V (4 blocks of 16 dims) ----
#pragma unroll
    for (int c = 0; c < 4; ++c) {
      const _Float16* vr = Vt + (size_t)(c * 16 + lrow) * NTOK + kt + (hi ? 16 : 0);
      accO[c] = __builtin_amdgcn_wmma_f32_16x16x32_f16(false, pa, false, load_frag_b(vr),
                                                       (short)0, accO[c], false, false);
    }
  }

  // ---- normalize and write O ----
#pragma unroll
  for (int c = 0; c < 4; ++c)
#pragma unroll
    for (int r = 0; r < 8; ++r) {
      int m = r + hi * 8;
      float v = accO[c][r] / lsum[r];
      obuf[(size_t)(b * NTOK + i0 + m) * DIM + h * DHD + c * 16 + lrow] = (_Float16)v;
    }
}

// ---------------------------------------------------------------------------
// host orchestration
// ---------------------------------------------------------------------------
static void launch_gemm(const _Float16* A, const _Float16* Bt, const float* bias,
                        const float* residual, float* Cf, _Float16* Ch,
                        int M, int N, int K, int gelu, hipStream_t s) {
  dim3 grid((N + 16 * GN - 1) / (16 * GN), (M + 16 * GM - 1) / (16 * GM));
  gemm_wmma_kernel<<<grid, dim3(32), 0, s>>>(A, Bt, bias, residual, Cf, Ch, M, N, K, gelu);
}

extern "C" void kernel_launch(void* const* d_in, const int* in_sizes, int n_in,
                              void* d_out, int out_size, void* d_ws, size_t ws_size,
                              hipStream_t stream) {
  (void)in_sizes; (void)n_in; (void)out_size; (void)ws_size;

  const float* x        = (const float*)d_in[0];
  const float* patch_w  = (const float*)d_in[1];
  const float* patch_b  = (const float*)d_in[2];
  const float* ln1_g    = (const float*)d_in[3];
  const float* ln1_b    = (const float*)d_in[4];
  const float* w_qkv    = (const float*)d_in[5];
  const float* bias_att = (const float*)d_in[6];
  const float* w_out    = (const float*)d_in[7];
  const float* ln2_g    = (const float*)d_in[8];
  const float* ln2_b    = (const float*)d_in[9];
  const float* w1       = (const float*)d_in[10];
  const float* b1       = (const float*)d_in[11];
  const float* w2       = (const float*)d_in[12];
  const float* b2       = (const float*)d_in[13];
  const float* lnh_g    = (const float*)d_in[14];
  const float* lnh_b    = (const float*)d_in[15];
  const float* w_head   = (const float*)d_in[16];
  const float* b_head   = (const float*)d_in[17];
  float* out = (float*)d_out;

  char* wp = (char*)d_ws;
  auto alloc = [&](size_t bytes) -> void* {
    void* r = (void*)wp;
    wp += (bytes + 255) & ~(size_t)255;
    return r;
  };

  _Float16* Apatch = (_Float16*)alloc((size_t)TOKS * PATCH_DIM * 2);
  _Float16* pwt    = (_Float16*)alloc((size_t)DIM * PATCH_DIM * 2);
  float*    tok    = (float*)   alloc((size_t)TOKS * DIM * 4);
  _Float16* hbuf   = (_Float16*)alloc((size_t)TOKS * DIM * 2);
  _Float16* qkvbuf = (_Float16*)alloc((size_t)TOKS * 3 * DIM * 2);
  _Float16* vtbuf  = (_Float16*)alloc((size_t)BATCH * HEADS * DHD * NTOK * 2);
  float*    sblog  = (float*)   alloc((size_t)HEADS * NTOK * NTOK * 4);
  _Float16* obuf   = (_Float16*)alloc((size_t)TOKS * DIM * 2);
  _Float16* mlpbuf = (_Float16*)alloc((size_t)TOKS * MLPD * 2);
  _Float16* wqkvt  = (_Float16*)alloc((size_t)3 * DIM * DIM * 2);
  _Float16* woutt  = (_Float16*)alloc((size_t)DIM * DIM * 2);
  _Float16* w1t    = (_Float16*)alloc((size_t)MLPD * DIM * 2);
  _Float16* w2t    = (_Float16*)alloc((size_t)DIM * MLPD * 2);
  _Float16* whdt   = (_Float16*)alloc((size_t)NCLS * DIM * 2);
  float*    pooled = (float*)   alloc((size_t)BATCH * DIM * 4);
  _Float16* ph     = (_Float16*)alloc((size_t)BATCH * DIM * 2);

  // --- patch embed ---
  {
    size_t tot = (size_t)TOKS * PATCH_DIM;
    im2col_kernel<<<(unsigned)((tot + 255) / 256), 256, 0, stream>>>(x, Apatch);
    size_t wtot = (size_t)PATCH_DIM * DIM;
    convt_kernel<<<(unsigned)((wtot + 255) / 256), 256, 0, stream>>>(patch_w, pwt, PATCH_DIM, DIM);
    launch_gemm(Apatch, pwt, patch_b, nullptr, tok, nullptr, TOKS, DIM, PATCH_DIM, 0, stream);
  }

  // --- transformer blocks ---
  for (int l = 0; l < DEPTH; ++l) {
    // LN1
    layernorm_kernel<<<TOKS, 256, 0, stream>>>(tok, ln1_g + l * DIM, ln1_b + l * DIM, hbuf, DIM);
    // QKV
    {
      size_t wtot = (size_t)DIM * 3 * DIM;
      convt_kernel<<<(unsigned)((wtot + 255) / 256), 256, 0, stream>>>(
          w_qkv + (size_t)l * DIM * 3 * DIM, wqkvt, DIM, 3 * DIM);
      launch_gemm(hbuf, wqkvt, nullptr, nullptr, nullptr, qkvbuf, TOKS, 3 * DIM, DIM, 0, stream);
    }
    // V transpose + bias log-softmax
    {
      size_t vtot = (size_t)BATCH * HEADS * DHD * NTOK;
      vtrans_kernel<<<(unsigned)((vtot + 255) / 256), 256, 0, stream>>>(qkvbuf, vtbuf);
      bias_logsoftmax_kernel<<<HEADS * NTOK, 256, 0, stream>>>(
          bias_att + (size_t)l * HEADS * NTOK * NTOK, sblog);
    }
    // attention
    attn_kernel<<<dim3(NTOK / 16, HEADS, BATCH), 32, 0, stream>>>(qkvbuf, vtbuf, sblog, obuf);
    // out projection + residual (in place on tok)
    {
      size_t wtot = (size_t)DIM * DIM;
      convt_kernel<<<(unsigned)((wtot + 255) / 256), 256, 0, stream>>>(
          w_out + (size_t)l * DIM * DIM, woutt, DIM, DIM);
      launch_gemm(obuf, woutt, nullptr, tok, tok, nullptr, TOKS, DIM, DIM, 0, stream);
    }
    // LN2 + MLP
    layernorm_kernel<<<TOKS, 256, 0, stream>>>(tok, ln2_g + l * DIM, ln2_b + l * DIM, hbuf, DIM);
    {
      size_t wtot = (size_t)DIM * MLPD;
      convt_kernel<<<(unsigned)((wtot + 255) / 256), 256, 0, stream>>>(
          w1 + (size_t)l * DIM * MLPD, w1t, DIM, MLPD);
      launch_gemm(hbuf, w1t, b1 + l * MLPD, nullptr, nullptr, mlpbuf, TOKS, MLPD, DIM, 1, stream);
      convt_kernel<<<(unsigned)((wtot + 255) / 256), 256, 0, stream>>>(
          w2 + (size_t)l * MLPD * DIM, w2t, MLPD, DIM);
      launch_gemm(mlpbuf, w2t, b2 + l * DIM, tok, tok, nullptr, TOKS, DIM, MLPD, 0, stream);
    }
  }

  // --- pool, head LN, classifier ---
  pool_kernel<<<BATCH, 256, 0, stream>>>(tok, pooled);
  layernorm_kernel<<<BATCH, 256, 0, stream>>>(pooled, lnh_g, lnh_b, ph, DIM);
  {
    size_t wtot = (size_t)DIM * NCLS;
    convt_kernel<<<(unsigned)((wtot + 255) / 256), 256, 0, stream>>>(w_head, whdt, DIM, NCLS);
    launch_gemm(ph, whdt, b_head, nullptr, out, nullptr, BATCH, NCLS, DIM, 0, stream);
  }
}
